// InfoNCELoss_19610820673889
// MI455X (gfx1250) — matrix operand order repaired and checked by reading, not compiled, not run on hardware
//
#include <hip/hip_runtime.h>
#include <math.h>
#include <stdint.h>

typedef __attribute__((ext_vector_type(16))) _Float16 v16h;
typedef __attribute__((ext_vector_type(8)))  _Float16 v8h;
typedef __attribute__((ext_vector_type(8)))  float    v8f;

#define NI   4096
#define NE   16384
#define ND   64
#define WPR  (NE / 32)          // 512 mask words per row
#define TINV (1.0f / 0.07f)

// ---------------- threefry-2x32-20 (exact JAX layout) ----------------
__device__ __forceinline__ uint32_t rotl32(uint32_t v, int r) { return (v << r) | (v >> (32 - r)); }

__device__ __forceinline__ void threefry2x32(uint32_t k0, uint32_t k1,
                                             uint32_t x0, uint32_t x1,
                                             uint32_t& o0, uint32_t& o1)
{
  const uint32_t k2 = k0 ^ k1 ^ 0x1BD11BDAu;
  x0 += k0; x1 += k1;
#define TF_R(r) { x0 += x1; x1 = rotl32(x1, r); x1 ^= x0; }
  TF_R(13) TF_R(15) TF_R(26) TF_R(6)   x0 += k1; x1 += k2 + 1u;
  TF_R(17) TF_R(29) TF_R(16) TF_R(24)  x0 += k2; x1 += k0 + 2u;
  TF_R(13) TF_R(15) TF_R(26) TF_R(6)   x0 += k0; x1 += k1 + 3u;
  TF_R(17) TF_R(29) TF_R(16) TF_R(24)  x0 += k1; x1 += k2 + 4u;
  TF_R(13) TF_R(15) TF_R(26) TF_R(6)   x0 += k2; x1 += k0 + 5u;
#undef TF_R
  o0 = x0; o1 = x1;
}

// Generate 23-bit comparison keys for jax.random.uniform(key(42), (NI,NE)).
// JAX (classic threefry path): counts = iota(n); pairs (j, j+n/2); bits[j]=out0,
// bits[j+n/2]=out1; u = bitcast((bits>>9)|0x3f800000)-1 -> ordering == (bits>>9).
__global__ void k_ukeys(uint32_t* __restrict__ ukey)
{
  const uint32_t h = (uint32_t)((size_t)NI * NE / 2);
  const uint32_t nthr = gridDim.x * blockDim.x;
  for (uint32_t j = blockIdx.x * blockDim.x + threadIdx.x; j < h; j += nthr) {
    uint32_t o0, o1;
    threefry2x32(0u, 42u, j, h + j, o0, o1);
    ukey[j]     = o0 >> 9;
    ukey[h + j] = o1 >> 9;
  }
}

// ---------------- row normalization (one wave per row, D=64) ----------------
__global__ void k_norm_item(const float* __restrict__ x, _Float16* __restrict__ y)
{
  const int lane = threadIdx.x & 31, wv = threadIdx.x >> 5;
  const int row = blockIdx.x * 8 + wv;
  float a = x[(size_t)row * ND + lane];
  float b = x[(size_t)row * ND + 32 + lane];
  float ss = a * a + b * b;
  for (int o = 16; o; o >>= 1) ss += __shfl_xor(ss, o, 32);
  float inv = 1.0f / sqrtf(ss);
  y[(size_t)row * ND + lane]      = (_Float16)(a * inv);
  y[(size_t)row * ND + 32 + lane] = (_Float16)(b * inv);
}

// entity: write transposed entT[d][e] so WMMA B fragments are contiguous loads
__global__ void k_norm_ent(const float* __restrict__ x, _Float16* __restrict__ yT)
{
  const int lane = threadIdx.x & 31, wv = threadIdx.x >> 5;
  const int row = blockIdx.x * 8 + wv;
  float a = x[(size_t)row * ND + lane];
  float b = x[(size_t)row * ND + 32 + lane];
  float ss = a * a + b * b;
  for (int o = 16; o; o >>= 1) ss += __shfl_xor(ss, o, 32);
  float inv = 1.0f / sqrtf(ss);
  yT[(size_t)lane * NE + row]        = (_Float16)(a * inv);
  yT[(size_t)(32 + lane) * NE + row] = (_Float16)(b * inv);
}

// ---------------- per-row negative selection (radix select + tie pass) -------
__global__ void __launch_bounds__(256) k_row_select(
    const int* __restrict__ adj, const uint32_t* __restrict__ ukey,
    uint32_t* __restrict__ posm, uint32_t* __restrict__ sampm,
    int* __restrict__ row_npos, int* __restrict__ row_k)
{
  __shared__ uint32_t s_pos[WPR];      // row pos bitmask
  __shared__ uint32_t s_hist[4096];    // reused 4096-bin / 2048-bin histogram
  __shared__ int s_red[8];
  __shared__ int s_wcnt[8];
  __shared__ int s_scal[6];
  __shared__ int s_base;

  const int tid = threadIdx.x, lane = tid & 31, wv = tid >> 5;
  const int row = blockIdx.x;
  const size_t rowoff = (size_t)row * NE;

  // pass 0: pos bitmask + num_pos
  int cnt = 0;
  for (int c = 0; c < NE; c += 256) {
    int e = c + tid;
    int pos = adj[rowoff + e] > 0;
    uint32_t bw = (uint32_t)__ballot(pos);
    if (lane == 0) s_pos[(c >> 5) + wv] = bw;
    cnt += pos;
  }
  for (int o = 16; o; o >>= 1) cnt += __shfl_xor(cnt, o, 32);
  if (lane == 0) s_red[wv] = cnt;
  __syncthreads();
  if (tid == 0) {
    int np = 0; for (int w = 0; w < 8; ++w) np += s_red[w];
    int nn = NE - np;
    int k = np < nn ? np : nn;          // floor(np * 1.0) clamped to num_neg
    s_scal[0] = np; s_scal[1] = k;
    row_npos[row] = np; row_k[row] = k;
  }
  __syncthreads();
  for (int w = tid; w < WPR; w += 256) posm[(size_t)row * WPR + w] = s_pos[w];
  const int k = s_scal[1];
  if (k <= 0) {
    for (int w = tid; w < WPR; w += 256) sampm[(size_t)row * WPR + w] = 0u;
    return;
  }

  // pass 1: 12-bit bucket histogram over negative keys
  for (int b = tid; b < 4096; b += 256) s_hist[b] = 0;
  __syncthreads();
  for (int c = 0; c < NE; c += 256) {
    int e = c + tid;
    if (!((s_pos[e >> 5] >> (e & 31)) & 1u)) {
      uint32_t key = ukey[rowoff + e];
      atomicAdd(&s_hist[key >> 11], 1u);
    }
  }
  __syncthreads();
  if (tid == 0) {
    int cum = 0, b = 0;
    for (;; ++b) { int hh = (int)s_hist[b]; if (cum + hh >= k) break; cum += hh; }
    s_scal[2] = b; s_scal[3] = k - cum;   // kprime >= 1
  }
  __syncthreads();
  const int bucket = s_scal[2], kprime = s_scal[3];

  // pass 2: 11-bit histogram inside the winning bucket
  for (int b = tid; b < 2048; b += 256) s_hist[b] = 0;
  __syncthreads();
  for (int c = 0; c < NE; c += 256) {
    int e = c + tid;
    if (!((s_pos[e >> 5] >> (e & 31)) & 1u)) {
      uint32_t key = ukey[rowoff + e];
      if ((int)(key >> 11) == bucket) atomicAdd(&s_hist[key & 2047u], 1u);
    }
  }
  __syncthreads();
  if (tid == 0) {
    int cum = 0, l = 0;
    for (;; ++l) { int hh = (int)s_hist[l]; if (cum + hh >= kprime) break; cum += hh; }
    s_scal[4] = (bucket << 11) | l;
    s_scal[5] = kprime - cum;            // ties_needed >= 1
    s_base = 0;
  }
  __syncthreads();
  const uint32_t thresh = (uint32_t)s_scal[4];
  const int ties_needed = s_scal[5];

  // pass 3: samp = neg & (key<thresh), plus first ties_needed ties in index order
  for (int c = 0; c < NE; c += 256) {
    int e = c + tid;
    int neg = !((s_pos[e >> 5] >> (e & 31)) & 1u);
    uint32_t key = neg ? ukey[rowoff + e] : 0xFFFFFFFFu;
    int less = neg && (key < thresh);
    int tie  = neg && (key == thresh);
    uint32_t tb = (uint32_t)__ballot(tie);
    int lpre = __popc(tb & ((1u << lane) - 1u));
    if (lane == 0) s_wcnt[wv] = __popc(tb);
    __syncthreads();
    int off = s_base;
    for (int w = 0; w < wv; ++w) off += s_wcnt[w];
    int sel = less || (tie && (off + lpre) < ties_needed);
    uint32_t sb = (uint32_t)__ballot(sel);
    if (lane == 0) sampm[(size_t)row * WPR + (c >> 5) + wv] = sb;
    __syncthreads();
    if (tid == 0) { int t = 0; for (int w = 0; w < 8; ++w) t += s_wcnt[w]; s_base += t; }
    __syncthreads();
  }
}

// ---------------- WMMA sim + fused loss ----------------
// A (16x32 f16): lane M=lane&15, elems = two contiguous 8-half chunks per kstep.
// B (32x16 f16): lane N=lane&15, elems K=8*(lane>>4)+0..7, 16+8*(lane>>4)+0..7 from entT[d][e].
// C (16x16 f32): reg r -> M = r + 8*(lane>>4), N = lane&15.
__device__ __forceinline__ v8f sim_tile(const v16h& a0, const v16h& a1,
                                        const _Float16* __restrict__ brow0,
                                        const _Float16* __restrict__ brow1,
                                        int eb)
{
  v16h b0, b1;
  {
    for (int t = 0; t < 8; ++t) { b0[t] = brow0[(size_t)t * NE + eb]; b0[8 + t] = brow0[(size_t)(16 + t) * NE + eb]; b1[t] = brow1[(size_t)t * NE + eb]; b1[8 + t] = brow1[(size_t)(16 + t) * NE + eb]; }
  }
  v8f c = {};
  c = __builtin_amdgcn_wmma_f32_16x16x32_f16(false, a0, false, b0, (short)0, c, false, false);
  c = __builtin_amdgcn_wmma_f32_16x16x32_f16(false, a1, false, b1, (short)0, c, false, false);
  return c;
}

__global__ void __launch_bounds__(256) k_loss(
    const _Float16* __restrict__ itemN, const _Float16* __restrict__ entT,
    const uint32_t* __restrict__ posm, const uint32_t* __restrict__ sampm,
    const int* __restrict__ row_npos, const int* __restrict__ row_k,
    float* __restrict__ partials)
{
  __shared__ uint32_t s_mask[16 * WPR];   // 32KB, reused: samp then pos
  __shared__ float s_wm[8][16];
  __shared__ float s_wS[8][16];
  __shared__ float s_m[16];
  __shared__ float s_S[16];
  __shared__ int   s_valid[16];
  __shared__ float s_acc[8];

  const int tid = threadIdx.x, lane = tid & 31, wv = tid >> 5;
  const int hf = lane >> 4, nl = lane & 15;
  const int rb = blockIdx.x * 16;

  for (int w = tid; w < 16 * WPR; w += 256) s_mask[w] = sampm[(size_t)rb * WPR + w];
  if (tid < 16) s_valid[tid] = (row_npos[rb + tid] > 0) && (row_k[rb + tid] > 0);
  __syncthreads();

  // A fragments for this block's 16 item rows (both k-steps), hoisted
  v16h a0, a1;
  {
    const v8h* p = (const v8h*)(itemN + (size_t)(rb + nl) * ND);
    v8h c1 = p[hf], c2 = p[2 + hf], d1 = p[4 + hf], d2 = p[6 + hf];
    for (int t = 0; t < 8; ++t) { a0[t] = c1[t]; a0[8 + t] = c2[t]; a1[t] = d1[t]; a1[8 + t] = d2[t]; }
  }
  const _Float16* brow0 = entT + (size_t)(8 * hf) * NE + nl;       // N = nl, K base 8*hf (d 0..31)
  const _Float16* brow1 = entT + (size_t)(32 + 8 * hf) * NE + nl;  // N = nl, K base 32+8*hf

  // Pass 1: streaming logsumexp over sampled negatives
  float m[8], s[8];
  for (int r = 0; r < 8; ++r) { m[r] = -INFINITY; s[r] = 0.0f; }
  for (int t = wv; t < NE / 16; t += 8) {
    const int eb = t * 16;
    v8f c = sim_tile(a0, a1, brow0, brow1, eb);
    const int e = eb + nl, widx = e >> 5, bsh = e & 31;
    for (int r = 0; r < 8; ++r) {
      const int M = hf * 8 + r;
      if ((s_mask[M * WPR + widx] >> bsh) & 1u) {
        float x = c[r] * TINV;
        if (x > m[r]) { s[r] = s[r] * expf(m[r] - x) + 1.0f; m[r] = x; }
        else          { s[r] += expf(x - m[r]); }
      }
    }
  }
  // combine across the 16 lanes of each half (xor stays within half)
  for (int o = 1; o < 16; o <<= 1) {
    for (int r = 0; r < 8; ++r) {
      float mo = __shfl_xor(m[r], o, 32);
      float so = __shfl_xor(s[r], o, 32);
      float M2 = fmaxf(m[r], mo);
      if (M2 == -INFINITY) s[r] = 0.0f;
      else s[r] = s[r] * expf(m[r] - M2) + so * expf(mo - M2);
      m[r] = M2;
    }
  }
  if (nl == 0) {
    for (int r = 0; r < 8; ++r) { s_wm[wv][hf * 8 + r] = m[r]; s_wS[wv][hf * 8 + r] = s[r]; }
  }
  __syncthreads();
  if (tid < 16) {
    float mm = -INFINITY, ss = 0.0f;
    for (int w = 0; w < 8; ++w) {
      float mo = s_wm[w][tid], so = s_wS[w][tid];
      float M2 = fmaxf(mm, mo);
      if (M2 == -INFINITY) ss = 0.0f;
      else ss = ss * expf(mm - M2) + so * expf(mo - M2);
      mm = M2;
    }
    s_m[tid] = mm; s_S[tid] = ss;
  }
  __syncthreads();
  // swap bitmask to positives
  for (int w = tid; w < 16 * WPR; w += 256) s_mask[w] = posm[(size_t)rb * WPR + w];
  __syncthreads();

  // Pass 2: recompute sim tiles, accumulate positive-pair CE losses
  float acc = 0.0f;
  for (int t = wv; t < NE / 16; t += 8) {
    const int eb = t * 16;
    v8f c = sim_tile(a0, a1, brow0, brow1, eb);
    const int e = eb + nl, widx = e >> 5, bsh = e & 31;
    for (int r = 0; r < 8; ++r) {
      const int M = hf * 8 + r;
      if (s_valid[M] && ((s_mask[M * WPR + widx] >> bsh) & 1u)) {
        float x  = c[r] * TINV;
        float mm = s_m[M], SS = s_S[M];
        float pm = fmaxf(x, mm);
        acc += logf(expf(x - pm) + SS * expf(mm - pm)) + pm - x;
      }
    }
  }
  for (int o = 16; o; o >>= 1) acc += __shfl_xor(acc, o, 32);
  if (lane == 0) s_acc[wv] = acc;
  __syncthreads();
  if (tid == 0) {
    float tot = 0.0f;
    for (int w = 0; w < 8; ++w) tot += s_acc[w];
    partials[blockIdx.x] = tot;
  }
}

// ---------------- final deterministic reduction ----------------
__global__ void k_final(const float* __restrict__ partials,
                        const int* __restrict__ row_npos, const int* __restrict__ row_k,
                        float* __restrict__ out)
{
  __shared__ float sf[256];
  __shared__ int   si[256];
  const int tid = threadIdx.x;
  sf[tid] = partials[tid];
  int np = 0;
  for (int r = tid; r < NI; r += 256) {
    int n = row_npos[r], k = row_k[r];
    if (n > 0 && k > 0) np += n;
  }
  si[tid] = np;
  __syncthreads();
  for (int o = 128; o; o >>= 1) {
    if (tid < o) { sf[tid] += sf[tid + o]; si[tid] += si[tid + o]; }
    __syncthreads();
  }
  if (tid == 0) out[0] = (si[0] > 0) ? (sf[0] / (float)si[0]) : 0.0f;
}

// ---------------- host launcher ----------------
extern "C" void kernel_launch(void* const* d_in, const int* in_sizes, int n_in,
                              void* d_out, int out_size, void* d_ws, size_t ws_size,
                              hipStream_t stream)
{
  (void)in_sizes; (void)n_in; (void)out_size; (void)ws_size;
  const float* item = (const float*)d_in[0];
  const float* ent  = (const float*)d_in[1];
  const int*   adj  = (const int*)d_in[2];
  float* out = (float*)d_out;

  char* ws = (char*)d_ws;
  size_t o = 0;
  uint32_t* ukey  = (uint32_t*)(ws + o); o += (size_t)NI * NE * 4;   // 256 MB keys
  uint32_t* posm  = (uint32_t*)(ws + o); o += (size_t)NI * WPR * 4;  // 8 MB
  uint32_t* sampm = (uint32_t*)(ws + o); o += (size_t)NI * WPR * 4;  // 8 MB
  _Float16* itemN = (_Float16*)(ws + o); o += (size_t)NI * ND * 2;
  _Float16* entT  = (_Float16*)(ws + o); o += (size_t)NE * ND * 2;
  int* row_npos   = (int*)(ws + o);      o += (size_t)NI * 4;
  int* row_k      = (int*)(ws + o);      o += (size_t)NI * 4;
  float* partials = (float*)(ws + o);    o += 256 * 4;

  k_ukeys     <<<32768,  256, 0, stream>>>(ukey);
  k_norm_item <<<NI / 8, 256, 0, stream>>>(item, itemN);
  k_norm_ent  <<<NE / 8, 256, 0, stream>>>(ent, entT);
  k_row_select<<<NI,     256, 0, stream>>>(adj, ukey, posm, sampm, row_npos, row_k);
  k_loss      <<<NI / 16,256, 0, stream>>>(itemN, entT, posm, sampm, row_npos, row_k, partials);
  k_final     <<<1,      256, 0, stream>>>(partials, row_npos, row_k, out);
}
